// SlotAttentionV2_28235115004082
// MI455X (gfx1250) — matrix operand order repaired and checked by reading, not compiled, not run on hardware
//
#include <hip/hip_runtime.h>
#include <hip/hip_bf16.h>

#define B_    256
#define N_    4096
#define NS_   8
#define D_    64
#define HID_  128
#define MH_   128
#define ITERS_ 3
#define EPS_   1e-8f
#define SCALE_ 0.125f   // 64^-0.5

typedef __attribute__((ext_vector_type(16))) __bf16 v16bf;
typedef __attribute__((ext_vector_type(8)))  __bf16 v8bf;
typedef __attribute__((ext_vector_type(8)))  float  v8f;

__device__ __forceinline__ float sigmoidf_(float x){ return 1.0f/(1.0f+__expf(-x)); }

// ---------------------------------------------------------------------------
// K0: reparameterized slot init + prior MLP + kl=0.   grid=2048, block=128
// ---------------------------------------------------------------------------
__global__ __launch_bounds__(128) void init_kernel(
    const float* __restrict__ slots_in, const float* __restrict__ prior_slots,
    const float* __restrict__ eps_noise,
    const float* __restrict__ sr_W1, const float* __restrict__ sr_b1,
    const float* __restrict__ sr_W2, const float* __restrict__ sr_b2,
    const float* __restrict__ pr_W1, const float* __restrict__ pr_b1,
    const float* __restrict__ pr_W2, const float* __restrict__ pr_b2,
    float* __restrict__ slots0, float* __restrict__ prior_out,
    float* __restrict__ klbuf)
{
  const int r = blockIdx.x; const int t = threadIdx.x;
  __shared__ float row[D_];
  __shared__ float h[HID_];
  __shared__ float ov[2*D_];
  if (t < D_) row[t] = slots_in[r*D_+t];
  __syncthreads();
  { float a = sr_b1[t];
    #pragma unroll 4
    for (int e=0;e<D_;++e) a += row[e]*sr_W1[t*D_+e];
    h[t] = fmaxf(a,0.f); }
  __syncthreads();
  { float a = sr_b2[t];
    #pragma unroll 4
    for (int e=0;e<HID_;++e) a += h[e]*sr_W2[t*HID_+e];
    ov[t] = a; }
  __syncthreads();
  if (t < D_) slots0[r*D_+t] = eps_noise[r*D_+t]*__expf(0.5f*ov[D_+t]) + ov[t];
  __syncthreads();
  if (t < D_) row[t] = prior_slots[r*D_+t];
  __syncthreads();
  { float a = pr_b1[t];
    #pragma unroll 4
    for (int e=0;e<D_;++e) a += row[e]*pr_W1[t*D_+e];
    h[t] = fmaxf(a,0.f); }
  __syncthreads();
  { float a = pr_b2[t];
    #pragma unroll 4
    for (int e=0;e<HID_;++e) a += h[e]*pr_W2[t*HID_+e];
    prior_out[r*2*D_+t] = a; }
  if (t == 0) klbuf[r] = 0.f;
}

// ---------------------------------------------------------------------------
// Kq: q = slots @ Wq^T + bq.   grid=2048, block=64
// ---------------------------------------------------------------------------
__global__ __launch_bounds__(64) void q_kernel(
    const float* __restrict__ slots, const float* __restrict__ Wq,
    const float* __restrict__ bq, float* __restrict__ qbuf)
{
  const int r = blockIdx.x; const int d = threadIdx.x;
  __shared__ float row[D_];
  row[d] = slots[r*D_+d];
  __syncthreads();
  float a = bq[d];
  #pragma unroll 4
  for (int e=0;e<D_;++e) a += row[e]*Wq[d*D_+e];
  qbuf[r*D_+d] = a;
}

// ---------------------------------------------------------------------------
// K2: fused LN + dots + slot-softmax + weighted-x accumulation (WMMA bf16).
// grid = B (one block per batch), block = 256 (8 waves, wave32).
// ---------------------------------------------------------------------------
__global__ __launch_bounds__(256) void attn_kernel(
    const float* __restrict__ inputs, const float* __restrict__ qbuf,
    const float* __restrict__ Wk, const float* __restrict__ bk,
    const float* __restrict__ ni_g, const float* __restrict__ ni_b,
    float* __restrict__ Pbuf, float* __restrict__ Sxbuf, float* __restrict__ Sbuf)
{
  __shared__ __align__(16) __bf16 qk[16*D_];        // scaled q@Wk, rows 8..15 zero
  __shared__ float qb[NS_];                         // scaled q.bk
  __shared__ float lng[D_], lnb[D_];
  __shared__ __align__(32) __bf16 xln[8*32*D_];     // per-wave 32x64 LN'd tile (bf16)
  __shared__ __align__(16) __bf16 smb[8*16*32];     // per-wave 16x32 softmax tile
  __shared__ float redP[8][9*D_];
  __shared__ float redS[8][NS_];

  const int b    = blockIdx.x;
  const int tid  = threadIdx.x;
  const int wave = tid >> 5;
  const int lane = tid & 31;

  const float* q = qbuf + b*NS_*D_;
  for (int o = tid; o < 16*D_; o += 256) {
    const int i = o >> 6, e = o & 63;
    float a = 0.f;
    if (i < NS_) {
      #pragma unroll 4
      for (int n=0;n<D_;++n) a += q[i*D_+n]*Wk[n*D_+e];
      a *= SCALE_;
    }
    qk[o] = (__bf16)a;
  }
  if (tid < NS_) {
    float a = 0.f;
    #pragma unroll 4
    for (int n=0;n<D_;++n) a += q[tid*D_+n]*bk[n];
    qb[tid] = a*SCALE_;
  }
  if (tid >= 64  && tid < 128) lng[tid-64]  = ni_g[tid-64];
  if (tid >= 128 && tid < 192) lnb[tid-128] = ni_b[tid-128];
  for (int o = lane; o < 8*32; o += 32) {
    const int rr = 8 + (o >> 5), cc = o & 31;
    smb[wave*512 + rr*32 + cc] = (rr == 8) ? (__bf16)1.0f : (__bf16)0.0f;
  }
  __syncthreads();

  v8f Pacc[4]; Pacc[0] = {}; Pacc[1] = {}; Pacc[2] = {}; Pacc[3] = {};
  float Sacc[8] = {0.f,0.f,0.f,0.f,0.f,0.f,0.f,0.f};

  const float* xb  = inputs + (size_t)b * N_ * D_;
  __bf16* myx = &xln[wave*32*D_];
  __bf16* mys = &smb[wave*512];
  const int m16 = lane & 15, g = lane >> 4;

  for (int step = 0; step < N_/256; ++step) {
    const int j = step*256 + wave*32 + lane;          // one input row per lane
    const float4* x4 = (const float4*)(xb + (size_t)j * D_);
    float s = 0.f, ss = 0.f;
    #pragma unroll
    for (int e=0;e<16;++e){
      float4 t4 = x4[e];
      s  += t4.x+t4.y+t4.z+t4.w;
      ss += t4.x*t4.x+t4.y*t4.y+t4.z*t4.z+t4.w*t4.w;
    }
    const float mean = s*(1.f/64.f);
    const float var  = ss*(1.f/64.f) - mean*mean;
    const float rstd = rsqrtf(var + 1e-5f);
    __bf16* xr = &myx[lane*D_];
    #pragma unroll
    for (int e=0;e<16;++e){
      float4 t4 = x4[e];
      const int c0 = e*4;
      xr[c0+0] = (__bf16)((t4.x-mean)*rstd*lng[c0+0]+lnb[c0+0]);
      xr[c0+1] = (__bf16)((t4.y-mean)*rstd*lng[c0+1]+lnb[c0+1]);
      xr[c0+2] = (__bf16)((t4.z-mean)*rstd*lng[c0+2]+lnb[c0+2]);
      xr[c0+3] = (__bf16)((t4.w-mean)*rstd*lng[c0+3]+lnb[c0+3]);
    }
    for (int hf=0; hf<2; ++hf){
      v8f dacc = {};
      #pragma unroll
      for (int c=0;c<2;++c){
        const __bf16* qr = &qk[m16*D_ + c*32 + g*8];
        v8bf alo = *(const v8bf*)qr;
        v8bf ahi = *(const v8bf*)(qr+16);
        v16bf afrag = __builtin_shufflevector(alo, ahi, 0,1,2,3,4,5,6,7,8,9,10,11,12,13,14,15);
        const __bf16* br = &myx[(hf*16+m16)*D_ + c*32 + g*8];
        v8bf blo = *(const v8bf*)br;
        v8bf bhi = *(const v8bf*)(br+16);
        v16bf bfrag = __builtin_shufflevector(blo, bhi, 0,1,2,3,4,5,6,7,8,9,10,11,12,13,14,15);
        dacc = __builtin_amdgcn_wmma_f32_16x16x32_bf16(false, afrag, false, bfrag,
                                                       (short)0, dacc, false, false);
      }
      float dv[8]; float mx = -1e30f;
      #pragma unroll
      for (int r2=0;r2<8;++r2){ dv[r2] = dacc[r2] + qb[r2]; mx = fmaxf(mx, dv[r2]); }
      float sum = 0.f;
      #pragma unroll
      for (int r2=0;r2<8;++r2){ dv[r2] = __expf(dv[r2]-mx); sum += dv[r2]; }
      const float inv = 1.f/sum;
      if (lane < 16){
        #pragma unroll
        for (int r2=0;r2<8;++r2){
          const float smv = dv[r2]*inv;
          Sacc[r2] += smv;
          mys[r2*32 + hf*16 + lane] = (__bf16)smv;
        }
      }
    }
    {
      const __bf16* sr = &mys[m16*32 + g*8];
      v8bf slo = *(const v8bf*)sr;
      v8bf shi = *(const v8bf*)(sr+16);
      v16bf sfrag = __builtin_shufflevector(slo, shi, 0,1,2,3,4,5,6,7,8,9,10,11,12,13,14,15);
      #pragma unroll
      for (int t2=0;t2<4;++t2){
        const int dcol = t2*16 + m16;
        const int jb   = g*8;
        v16bf bfrag;
        #pragma unroll
        for (int kk=0;kk<16;++kk) bfrag[kk] = myx[(jb+(kk&7)+(kk>>3)*16)*D_ + dcol];
        Pacc[t2] = __builtin_amdgcn_wmma_f32_16x16x32_bf16(false, sfrag, false, bfrag,
                                                           (short)0, Pacc[t2], false, false);
      }
    }
  }

  #pragma unroll
  for (int r2=0;r2<8;++r2){
    float v = Sacc[r2];
    for (int off=16; off>=1; off>>=1) v += __shfl_xor(v, off, 32);
    if (lane == 0) redS[wave][r2] = v;
  }
  #pragma unroll
  for (int t2=0;t2<4;++t2){
    const int dcol = t2*16 + m16;
    #pragma unroll
    for (int r2=0;r2<8;++r2){
      const int irow = r2 + g*8;
      if (irow < 9) redP[wave][irow*D_ + dcol] = Pacc[t2][r2];
    }
  }
  __syncthreads();
  for (int o = tid; o < 9*D_; o += 256){
    float sum = 0.f;
    #pragma unroll
    for (int w=0;w<8;++w) sum += redP[w][o];
    const int irow = o >> 6, dcol = o & 63;
    if (irow < 8) Pbuf[b*NS_*D_ + o] = sum;
    else          Sxbuf[b*D_ + dcol] = sum;
  }
  if (tid < NS_){
    float sum = 0.f;
    #pragma unroll
    for (int w=0;w<8;++w) sum += redS[w][tid];
    Sbuf[b*NS_+tid] = sum;
  }
}

// ---------------------------------------------------------------------------
// K3: updates -> GRU -> LN+MLP residual -> posterior -> KL.  grid=2048, block=64
// ---------------------------------------------------------------------------
__global__ __launch_bounds__(64) void update_kernel(
    const float* __restrict__ slots_prev,
    const float* __restrict__ Pbuf, const float* __restrict__ Sxbuf,
    const float* __restrict__ Sbuf, const float* __restrict__ prior,
    const float* __restrict__ Wv,  const float* __restrict__ bv,
    const float* __restrict__ gWih, const float* __restrict__ gWhh,
    const float* __restrict__ gbih, const float* __restrict__ gbhh,
    const float* __restrict__ npf_g, const float* __restrict__ npf_b,
    const float* __restrict__ mW1, const float* __restrict__ mb1,
    const float* __restrict__ mW2, const float* __restrict__ mb2,
    const float* __restrict__ sW1, const float* __restrict__ sb1,
    const float* __restrict__ sW2, const float* __restrict__ sb2,
    float* __restrict__ slots_next, float* __restrict__ klbuf)
{
  const int r = blockIdx.x; const int b = r >> 3; const int i = r & 7;
  const int d = threadIdx.x;
  __shared__ float uvec[D_], upd[D_], gi[3*D_], gh[3*D_], hvec[D_];
  __shared__ float snew[D_], lnv[D_], hid[HID_], post[2*D_], red[D_];

  const float den = Sbuf[b*NS_+i] + EPS_*(float)N_;
  uvec[d] = (Pbuf[r*D_+d] + EPS_*Sxbuf[b*D_+d]) / den;
  hvec[d] = slots_prev[r*D_+d];
  __syncthreads();
  { float a = bv[d];
    #pragma unroll 4
    for (int e=0;e<D_;++e) a += uvec[e]*Wv[d*D_+e];
    upd[d] = a; }
  __syncthreads();
  for (int g3=0; g3<3; ++g3){
    const int o2 = g3*D_ + d;
    float a = gbih[o2]; float h2 = gbhh[o2];
    #pragma unroll 4
    for (int e=0;e<D_;++e){ a += upd[e]*gWih[o2*D_+e]; h2 += hvec[e]*gWhh[o2*D_+e]; }
    gi[o2] = a; gh[o2] = h2;
  }
  __syncthreads();
  {
    const float rg = sigmoidf_(gi[d]      + gh[d]);
    const float zg = sigmoidf_(gi[D_+d]   + gh[D_+d]);
    const float ng = tanhf   (gi[2*D_+d] + rg*gh[2*D_+d]);
    snew[d] = (1.f-zg)*ng + zg*hvec[d];
  }
  __syncthreads();
  { float sm = 0.f, sq = 0.f;
    #pragma unroll 4
    for (int e=0;e<D_;++e){ const float v = snew[e]; sm += v; sq += v*v; }
    const float mean = sm*(1.f/64.f);
    const float var  = sq*(1.f/64.f) - mean*mean;
    const float rstd = rsqrtf(var + 1e-5f);
    lnv[d] = (snew[d]-mean)*rstd*npf_g[d] + npf_b[d]; }
  __syncthreads();
  for (int h2=0; h2<2; ++h2){
    const int o2 = h2*D_ + d; float a = mb1[o2];
    #pragma unroll 4
    for (int e=0;e<D_;++e) a += lnv[e]*mW1[o2*D_+e];
    hid[o2] = fmaxf(a,0.f);
  }
  __syncthreads();
  float sf;
  { float a = mb2[d];
    #pragma unroll 4
    for (int e=0;e<MH_;++e) a += hid[e]*mW2[d*MH_+e];
    sf = snew[d] + a;
    slots_next[r*D_+d] = sf; }
  __syncthreads();
  snew[d] = sf;
  __syncthreads();
  for (int h2=0; h2<2; ++h2){
    const int o2 = h2*D_ + d; float a = sb1[o2];
    #pragma unroll 4
    for (int e=0;e<D_;++e) a += snew[e]*sW1[o2*D_+e];
    hid[o2] = fmaxf(a,0.f);
  }
  __syncthreads();
  for (int p2=0; p2<2; ++p2){
    const int o2 = p2*D_ + d; float a = sb2[o2];
    #pragma unroll 4
    for (int e=0;e<HID_;++e) a += hid[e]*sW2[o2*HID_+e];
    post[o2] = a;
  }
  __syncthreads();
  { const float m_po = post[d],         lv_po = post[D_+d];
    const float m_pr = prior[r*2*D_+d], lv_pr = prior[r*2*D_+D_+d];
    const float dm = m_po - m_pr;
    red[d] = 0.5f*(lv_pr - lv_po + (__expf(lv_po)+dm*dm)*__expf(-lv_pr) - 1.f); }
  __syncthreads();
  if (d == 0){
    float s = 0.f;
    #pragma unroll 4
    for (int e=0;e<D_;++e) s += red[e];
    klbuf[r] += s;
  }
}

// ---------------------------------------------------------------------------
__global__ void final_kernel(const float* __restrict__ slots,
                             const float* __restrict__ kl,
                             float* __restrict__ out)
{
  const int i = blockIdx.x*blockDim.x + threadIdx.x;
  const int NSLOT = B_*NS_*D_;
  if (i < NSLOT) out[i] = slots[i];
  else if (i < NSLOT + B_*NS_) out[i] = kl[i - NSLOT];
}

// ---------------------------------------------------------------------------
extern "C" void kernel_launch(void* const* d_in, const int* in_sizes, int n_in,
                              void* d_out, int out_size, void* d_ws, size_t ws_size,
                              hipStream_t stream) {
  const float* inputs      = (const float*)d_in[0];
  const float* slots_in    = (const float*)d_in[1];
  const float* prior_slots = (const float*)d_in[2];
  const float* eps_noise   = (const float*)d_in[3];
  const float* ni_g  = (const float*)d_in[4];
  const float* ni_b  = (const float*)d_in[5];
  const float* npf_g = (const float*)d_in[6];
  const float* npf_b = (const float*)d_in[7];
  const float* Wq = (const float*)d_in[8];   const float* bq = (const float*)d_in[9];
  const float* Wk = (const float*)d_in[10];  const float* bk = (const float*)d_in[11];
  const float* Wv = (const float*)d_in[12];  const float* bv = (const float*)d_in[13];
  const float* sr_W1 = (const float*)d_in[14]; const float* sr_b1 = (const float*)d_in[15];
  const float* sr_W2 = (const float*)d_in[16]; const float* sr_b2 = (const float*)d_in[17];
  const float* pr_W1 = (const float*)d_in[18]; const float* pr_b1 = (const float*)d_in[19];
  const float* pr_W2 = (const float*)d_in[20]; const float* pr_b2 = (const float*)d_in[21];
  const float* gWih = (const float*)d_in[22]; const float* gWhh = (const float*)d_in[23];
  const float* gbih = (const float*)d_in[24]; const float* gbhh = (const float*)d_in[25];
  const float* mW1 = (const float*)d_in[26];  const float* mb1 = (const float*)d_in[27];
  const float* mW2 = (const float*)d_in[28];  const float* mb2 = (const float*)d_in[29];

  const size_t R = (size_t)B_*NS_;             // 2048 slot rows
  float* ws = (float*)d_ws;
  float* slotsA = ws;
  float* slotsB = slotsA + R*D_;
  float* prior  = slotsB + R*D_;
  float* qbuf   = prior  + R*2*D_;
  float* Pbuf   = qbuf   + R*D_;
  float* Sxbuf  = Pbuf   + R*D_;
  float* Sbuf   = Sxbuf  + (size_t)B_*D_;
  float* klbuf  = Sbuf   + (size_t)B_*NS_;
  const size_t need = (size_t)((klbuf + R) - ws) * sizeof(float);
  if (ws_size < need) return;

  init_kernel<<<(int)R, 128, 0, stream>>>(slots_in, prior_slots, eps_noise,
                                          sr_W1, sr_b1, sr_W2, sr_b2,
                                          pr_W1, pr_b1, pr_W2, pr_b2,
                                          slotsA, prior, klbuf);
  float* cur = slotsA; float* nxt = slotsB;
  for (int it = 0; it < ITERS_; ++it) {
    q_kernel<<<(int)R, 64, 0, stream>>>(cur, Wq, bq, qbuf);
    attn_kernel<<<B_, 256, 0, stream>>>(inputs, qbuf, Wk, bk, ni_g, ni_b,
                                        Pbuf, Sxbuf, Sbuf);
    update_kernel<<<(int)R, 64, 0, stream>>>(cur, Pbuf, Sxbuf, Sbuf, prior,
                                             Wv, bv, gWih, gWhh, gbih, gbhh,
                                             npf_g, npf_b, mW1, mb1, mW2, mb2,
                                             sr_W1, sr_b1, sr_W2, sr_b2,
                                             nxt, klbuf);
    float* t = cur; cur = nxt; nxt = t;
  }
  const int total = B_*NS_*D_ + B_*NS_;
  final_kernel<<<(total + 255)/256, 256, 0, stream>>>(cur, klbuf, (float*)d_out);
}
